// Branch_21045339750664
// MI455X (gfx1250) — matrix operand order repaired and checked by reading, not compiled, not run on hardware
//
#include <hip/hip_runtime.h>
#include <hip/hip_bf16.h>

typedef __attribute__((ext_vector_type(16))) _Float16 v16h;
typedef __attribute__((ext_vector_type(8)))  float    v8f;
typedef __attribute__((ext_vector_type(4)))  float    v4f;

union HV  { v16h h; v4f f[2]; };   // assemble a 16-half WMMA fragment from two 16B loads
union H8  { v4f f; _Float16 h[8]; };

#define ASTRIDE 2320               // A-tile row stride in halves (2304 + 16 pad)

// ---------------------------------------------------------------------------
// Pack weight [256,256,3,3] f32 -> f16 B fragments laid out so each lane of a
// wave fetches its WMMA B operand with two contiguous b128 loads.
// Fragment layout (per ISA 05_wmma 16-bit B 32x16): lanes 0-15 hold K=0..7,16..23
// (N = lane), lanes 16-31 hold K=8..15,24..31 (N = lane-16); halves packed K-major.
// Global k index: k = tap*256 + c  (tap = ky*3+kx), matching the A-tile.
// ---------------------------------------------------------------------------
__global__ void pack_b_kernel(const float* __restrict__ w, _Float16* __restrict__ bp) {
  int idx = blockIdx.x * 256 + threadIdx.x;          // 72*16*2*256 = 589824 total
  int j    =  idx        & 7;
  int lane = (idx >> 3)  & 31;
  int h    = (idx >> 8)  & 1;
  int nt   = (idx >> 9)  & 15;
  int ks   =  idx >> 13;
  int N  = nt * 16 + (lane & 15);
  int kk = ks * 32 + ((lane < 16) ? 0 : 8) + h * 16 + j;
  int tap = kk >> 8;
  int c   = kk & 255;
  bp[idx] = (_Float16)w[(N * 256 + c) * 9 + tap];
}

// ---------------------------------------------------------------------------
// Offset conv (3x3, zero pad): off18[b, start+pix, p] for p in 0..17.
// Offset weights (18*2304 f32 = 162 KB) staged once per block in LDS.
// Block = 256 threads (thread = channel), 64 pixels per block.
// ---------------------------------------------------------------------------
__global__ void offsets_kernel(const float* __restrict__ inp,
                               const float* __restrict__ ow,
                               const float* __restrict__ ob,
                               float* __restrict__ offs,
                               int H, int W, int start, int total) {
  extern __shared__ float smemf[];
  float* lw  = smemf;            // 41472 floats
  float* red = smemf + 41472;    // 18 floats
  const int tid = threadIdx.x;
  for (int i = tid; i < 41472; i += 256) lw[i] = ow[i];
  __syncthreads();

  const int HW = H * W;
  const int M  = 4 * HW;
  const int c  = tid;
  for (int mi = 0; mi < 64; ++mi) {
    int m = blockIdx.x * 64 + mi;
    if (m >= M) break;                       // uniform across block
    int b = m / HW, pix = m % HW;
    int y = pix / W, x = pix % W;
    float acc[18];
#pragma unroll
    for (int p = 0; p < 18; ++p) acc[p] = 0.f;
#pragma unroll
    for (int t = 0; t < 9; ++t) {
      int yy = y + t / 3 - 1, xx = x + t % 3 - 1;
      if (yy < 0 || yy >= H || xx < 0 || xx >= W) continue;   // zero pad
      float v = inp[((size_t)b * total + start + yy * W + xx) * 256 + c];
#pragma unroll
      for (int p = 0; p < 18; ++p) acc[p] += lw[p * 2304 + c * 9 + t] * v;
    }
    // wave32 reduction, then cross-wave LDS atomic reduce
#pragma unroll
    for (int p = 0; p < 18; ++p) {
      float a = acc[p];
      for (int o = 16; o > 0; o >>= 1) a += __shfl_xor(a, o, 32);
      acc[p] = a;
    }
    if (tid < 18) red[tid] = 0.f;
    __syncthreads();
    if ((tid & 31) == 0) {
#pragma unroll
      for (int p = 0; p < 18; ++p) atomicAdd(&red[p], acc[p]);
    }
    __syncthreads();
    if (tid < 18)
      offs[((size_t)b * total + start + pix) * 18 + tid] = red[tid] + ob[tid];
    __syncthreads();
  }
}

// ---------------------------------------------------------------------------
// Fused deformable-im2col + WMMA GEMM for one level.
// Block: 256 threads (8 waves), 32 output rows, full N=256.
// Phase 1: 8 waves sample 32 rows x 9 taps x 256 ch (bilinear) into a
//          [32][2304] f16 LDS tile (k = tap*256 + c).
// Phase 2: wave w owns 32 cols [w*32, w*32+32) and BOTH 16-row halves:
//          per K-step load 2 A frags (LDS b128) + 2 B frags (global b128),
//          issue 4x v_wmma_f32_16x16x32_f16. No duplicated B loads in block.
// ---------------------------------------------------------------------------
__global__ void deform_gemm_kernel(const float* __restrict__ inp,
                                   const float* __restrict__ offs,
                                   const _Float16* __restrict__ bp,
                                   float* __restrict__ out,
                                   int H, int W, int start, int total) {
  extern __shared__ _Float16 ldsA[];           // [32][ASTRIDE]
  const int tid  = threadIdx.x;
  const int lane = tid & 31;
  const int wv   = tid >> 5;
  const int HW   = H * W;
  const int Mlev = 4 * HW;
  const int m0   = blockIdx.x * 32;

  // ---- Phase 1: deformable bilinear sampling into LDS A-tile ----
  for (int pr = wv; pr < 288; pr += 8) {       // 32 rows x 9 taps
    int ri = pr / 9, t = pr % 9;
    int m  = m0 + ri;
    _Float16* dst = ldsA + ri * ASTRIDE + t * 256 + lane * 8;
    H8 u;
    if (m >= Mlev) {
#pragma unroll
      for (int j = 0; j < 8; ++j) u.h[j] = (_Float16)0.f;
      *reinterpret_cast<v4f*>(dst) = u.f;
      continue;
    }
    int b = m / HW, pix = m % HW;
    int y = pix / W, x = pix % W;
    const float* op = offs + ((size_t)b * total + start + pix) * 18;
    float dy = op[2 * t], dx = op[2 * t + 1];
    float ys = (float)(y + t / 3 - 1) + dy;
    float xs = (float)(x + t % 3 - 1) + dx;
    float y0 = floorf(ys), x0 = floorf(xs);
    float wy = ys - y0,    wx = xs - x0;
    int iy0 = (int)y0, ix0 = (int)x0;
    float v[8];
#pragma unroll
    for (int j = 0; j < 8; ++j) v[j] = 0.f;
    const float* base = inp + ((size_t)b * total + start) * 256 + lane * 8;
#pragma unroll
    for (int cn = 0; cn < 4; ++cn) {           // 4 bilinear corners
      int yy = iy0 + (cn >> 1), xx = ix0 + (cn & 1);
      if (yy < 0 || yy >= H || xx < 0 || xx >= W) continue;  // OOB weight = 0
      float wgt = ((cn >> 1) ? wy : 1.f - wy) * ((cn & 1) ? wx : 1.f - wx);
      const float* p = base + (size_t)(yy * W + xx) * 256;
#pragma unroll
      for (int j = 0; j < 8; ++j) v[j] += wgt * p[j];        // 2x b128 loads
    }
#pragma unroll
    for (int j = 0; j < 8; ++j) u.h[j] = (_Float16)v[j];
    *reinterpret_cast<v4f*>(dst) = u.f;        // ds_store_b128
  }
  __syncthreads();

  // ---- Phase 2: WMMA (wave = 32 cols x 32 rows) ----
  v8f acc00 = {}, acc01 = {}, acc10 = {}, acc11 = {};
  const int lrow = lane & 15;
  const int ksel = (lane < 16) ? 0 : 8;        // per ISA 16-bit A 16x32 layout
  const _Float16* arp0 = ldsA + lrow * ASTRIDE + ksel;         // rows 0..15
  const _Float16* arp1 = ldsA + (16 + lrow) * ASTRIDE + ksel;  // rows 16..31

  for (int ks = 0; ks < 72; ++ks) {
    const _Float16* ap0   = arp0 + ks * 32;
    const _Float16* ap1   = arp1 + ks * 32;
    const _Float16* bbase = bp + (size_t)((ks * 16 + wv * 2) * 2) * 256 + lane * 8;
    if (ks < 71) __builtin_prefetch(bbase + 8 * 256, 0, 3);    // packed B 8*256 halves ahead (n-tile +4)
    HV a0, a1, b0, b1;
    b0.f[0] = *reinterpret_cast<const v4f*>(bbase + 0 * 256);
    b0.f[1] = *reinterpret_cast<const v4f*>(bbase + 1 * 256);
    b1.f[0] = *reinterpret_cast<const v4f*>(bbase + 2 * 256);
    b1.f[1] = *reinterpret_cast<const v4f*>(bbase + 3 * 256);
    a0.f[0] = *reinterpret_cast<const v4f*>(ap0);              // K 0..7  (+ksel)
    a0.f[1] = *reinterpret_cast<const v4f*>(ap0 + 16);         // K 16..23 (+ksel)
    a1.f[0] = *reinterpret_cast<const v4f*>(ap1);
    a1.f[1] = *reinterpret_cast<const v4f*>(ap1 + 16);
    acc00 = __builtin_amdgcn_wmma_f32_16x16x32_f16(false, a0.h, false, b0.h,
                                                   (short)0, acc00, false, false);
    acc01 = __builtin_amdgcn_wmma_f32_16x16x32_f16(false, a0.h, false, b1.h,
                                                   (short)0, acc01, false, false);
    acc10 = __builtin_amdgcn_wmma_f32_16x16x32_f16(false, a1.h, false, b0.h,
                                                   (short)0, acc10, false, false);
    acc11 = __builtin_amdgcn_wmma_f32_16x16x32_f16(false, a1.h, false, b1.h,
                                                   (short)0, acc11, false, false);
  }

  // ---- Phase 3: store D (16x16 f32 layout: VGPR r -> M=r / r+8, N=lane%16) ----
#pragma unroll
  for (int q = 0; q < 4; ++q) {
    const v8f& ac = (q == 0) ? acc00 : (q == 1) ? acc01 : (q == 2) ? acc10 : acc11;
    int mh = q >> 1;                           // row half
    int i  = q & 1;                            // 16-col subtile within wave's 32
#pragma unroll
    for (int r = 0; r < 8; ++r) {
      int Ml = mh * 16 + r + ((lane < 16) ? 0 : 8);
      int m  = m0 + Ml;
      if (m < Mlev) {
        int b = m / HW, pix = m % HW;
        int ncol = wv * 32 + i * 16 + lrow;
        out[((size_t)b * total + start + pix) * 256 + ncol] = ac[r];
      }
    }
  }
}

// ---------------------------------------------------------------------------
static const int LH[4]     = {100, 50, 25, 13};
static const int LWD[4]    = {152, 76, 38, 19};
static const int LSTART[4] = {0, 15200, 19000, 19950};
#define TOTAL 20197

extern "C" void kernel_launch(void* const* d_in, const int* in_sizes, int n_in,
                              void* d_out, int out_size, void* d_ws, size_t ws_size,
                              hipStream_t stream) {
  const float* inputs = (const float*)d_in[0];
  // d_in[1] = spatial_shapes (static, hardcoded above)
  const float* weight = (const float*)d_in[2];
  const float* ow     = (const float*)d_in[3];
  const float* ob     = (const float*)d_in[4];
  float* out = (float*)d_out;

  // ws layout: offsets f32 [4][20197][18] (5,816,736 B, padded), then packed B f16
  float*    offs  = (float*)d_ws;
  _Float16* bpack = (_Float16*)((char*)d_ws + 5816832);

  pack_b_kernel<<<2304, 256, 0, stream>>>(weight, bpack);

  for (int l = 0; l < 4; ++l) {
    int H = LH[l], W = LWD[l], start = LSTART[l];
    int M = 4 * H * W;
    offsets_kernel<<<(M + 63) / 64, 256, (41472 + 32) * sizeof(float), stream>>>(
        inputs, ow, ob, offs, H, W, start, TOTAL);
    deform_gemm_kernel<<<(M + 31) / 32, 256, 32 * ASTRIDE * sizeof(_Float16), stream>>>(
        inputs, offs, bpack, out, H, W, start, TOTAL);
  }
}
